// PromptLearner_79070347919546
// MI455X (gfx1250) — matrix operand order repaired and checked by reading, not compiled, not run on hardware
//
#include <hip/hip_runtime.h>
#include <hip/hip_bf16.h>

// ---------------------------------------------------------------------------
// PromptLearner projection for MI455X (gfx1250, wave32, WMMA + async-to-LDS).
//
// argmin_v ||x - e_v||^2  ==  argmax_v ( x.e_v - 0.5*||e_v||^2 )
// Dominant cost: [8192x768]x[768x49408] GEMM (~0.62 TFLOP) -> compute bound
// (HBM ~8us at 23.3 TB/s; X=25MB, E=152MB both L2 resident in 192MB L2).
// Strategy: bf16 WMMA (v_wmma_f32_16x16x32_bf16) with fused argmax.
//   - A strip (16x768 bf16 per wave) pinned in 192 VGPRs, loaded once.
//   - B tiles (32 vocab rows x 768) double-buffered in LDS, filled by
//     GLOBAL_LOAD_ASYNC_TO_LDS_B128 (ASYNCcnt): global traffic fully
//     overlaps the 48-WMMA compute block between barrier pairs.
//   - Two independent accumulator chains (even/odd K) for WMMA ILP; B
//     register fragments software-pipelined one K-step ahead.
// ---------------------------------------------------------------------------

typedef __attribute__((ext_vector_type(16))) __bf16        v16bf;
typedef __attribute__((ext_vector_type(8)))  float         v8f;
typedef __attribute__((ext_vector_type(4)))  unsigned int  u32x4;

#define D_DIM    768
#define V_DIM    49408
#define N_CLS    512
#define N_CTX    16
#define SUF_LEN  60
#define SEQ_LEN  77
#define NROWS    (N_CLS * N_CTX)        // 8192
#define KB       (D_DIM / 32)           // 24 K-blocks of 32
#define SPLIT    8                      // V-dimension split for occupancy
#define NT_PER_SPLIT ((V_DIM / 32) / SPLIT)  // 193 32-wide tiles per split
#define M_PER_WG 128                    // 8 waves x 16 rows
#define LDBP     776                    // padded LDS row stride (ushorts): 388 dw, %64==4
#define BUF_USH  (32 * LDBP)            // one buffer: 32 vocab rows
#define BUF_BYTE (BUF_USH * 2)          // 49664 B per buffer
#define CHUNKS   12                     // 16B chunks per thread per 32-row tile

__device__ __forceinline__ unsigned short f32_to_bf16(float f) {
    unsigned int u = __float_as_uint(f);
    unsigned int r = u + 0x7FFFu + ((u >> 16) & 1u);   // round to nearest even
    return (unsigned short)(r >> 16);
}

union FragU { u32x4 u[2]; v16bf v; };

// Async copy 16B global -> LDS (ISA §15.18.3 op 98, tracked by ASYNCcnt).
// LDS byte address = low 32 bits of the flat shared pointer.
__device__ __forceinline__ void async_b128(unsigned lds_byte, const void* g) {
    asm volatile("global_load_async_to_lds_b128 %0, %1, off"
                 :: "v"(lds_byte), "v"((unsigned long long)(size_t)g)
                 : "memory");
}
__device__ __forceinline__ void wait_async_le12() {
    asm volatile("s_wait_asynccnt 0xc" ::: "memory");
}
__device__ __forceinline__ void wait_async_0() {
    asm volatile("s_wait_asynccnt 0x0" ::: "memory");
}

// ---------------------------------------------------------------------------
// Kernel 1: ctx f32 -> bf16
// ---------------------------------------------------------------------------
__global__ void cvt_x_kernel(const float* __restrict__ x,
                             unsigned short* __restrict__ xbf, int n) {
    int i = blockIdx.x * blockDim.x + threadIdx.x;
    if (i < n) xbf[i] = f32_to_bf16(x[i]);
}

// ---------------------------------------------------------------------------
// Kernel 2: token_embedding f32 -> bf16, plus 0.5*||e||^2 per row
// ---------------------------------------------------------------------------
__global__ void cvt_e_kernel(const float* __restrict__ e,
                             unsigned short* __restrict__ ebf,
                             float* __restrict__ halfe2) {
    __shared__ float red[256];
    const int v = blockIdx.x;
    const float* src = e + (size_t)v * D_DIM;
    unsigned short* dst = ebf + (size_t)v * D_DIM;
    float s = 0.f;
    for (int d = threadIdx.x; d < D_DIM; d += 256) {
        float f = src[d];
        dst[d] = f32_to_bf16(f);
        s += f * f;
    }
    red[threadIdx.x] = s;
    __syncthreads();
    for (int off = 128; off > 0; off >>= 1) {
        if (threadIdx.x < off) red[threadIdx.x] += red[threadIdx.x + off];
        __syncthreads();
    }
    if (threadIdx.x == 0) halfe2[v] = 0.5f * red[0];
}

// ---------------------------------------------------------------------------
// Kernel 3: fused GEMM + argmax.
//   grid = (64 M-blocks, 8 V-splits), block = 256 (8 waves).
// ---------------------------------------------------------------------------
__global__ void __launch_bounds__(256)
gemm_argmax_kernel(const unsigned short* __restrict__ xbf,
                   const unsigned short* __restrict__ ebf,
                   const float* __restrict__ halfe2,
                   float* __restrict__ partVal,
                   int*   __restrict__ partIdx) {
    __shared__ __align__(16) unsigned short Blds[2][BUF_USH];  // 2 x 49.6 KB

    const int tidx  = threadIdx.x;
    const int lane  = tidx & 31;
    const int wave  = tidx >> 5;
    const int half  = lane >> 4;     // selects K sub-blocks (ISA A/B layouts)
    const int nlane = lane & 15;     // column within 16-wide tile
    const int mBlk  = blockIdx.x;    // 0..63
    const int split = blockIdx.y;    // 0..7

    const int aRow0 = mBlk * M_PER_WG + wave * 16;

    // ---- per-thread async-copy geometry: 12 x 16B chunks of the 32x768 tile
    int goff[CHUNKS];
    unsigned lbyte[CHUNKS];
    {
        const unsigned b0 = (unsigned)(size_t)&Blds[0][0];
        #pragma unroll
        for (int c = 0; c < CHUNKS; ++c) {
            int chunk = tidx + c * 256;           // 0..3071
            int brow  = chunk / 96;               // 96 8-elem chunks per row
            int bcol  = (chunk - brow * 96) * 8;
            goff[c]   = brow * D_DIM + bcol;      // ushort offset in E block
            lbyte[c]  = b0 + (unsigned)((brow * LDBP + bcol) * 2);
        }
    }

    // ---- A strip for this wave: 24 K-fragments, contiguous-K A layout (same k order as B):
    //      lanes 0-15: row=lane, K = k0+0..15
    //      lanes 16-31: row=lane-16, K = k0+16..31
    v16bf afrag[KB];
    {
        const unsigned short* ap =
            xbf + (size_t)(aRow0 + nlane) * D_DIM + half * 16;
        #pragma unroll
        for (int kb = 0; kb < KB; ++kb) {
            FragU f;
            f.u[0] = *(const u32x4*)(ap + kb * 32);
            f.u[1] = *(const u32x4*)(ap + kb * 32 + 8);
            afrag[kb] = f.v;
        }
    }

    float bestv[8];
    int   besti[8];
    #pragma unroll
    for (int i = 0; i < 8; ++i) { bestv[i] = -3.4e38f; besti[i] = 0; }

    const int tile0 = split * NT_PER_SPLIT;

    // prologue: async-fill buffer 0 with tile 0 (32 vocab rows)
    {
        const unsigned short* src = ebf + (size_t)(tile0 * 32) * D_DIM;
        #pragma unroll
        for (int c = 0; c < CHUNKS; ++c) async_b128(lbyte[c], src + goff[c]);
    }

    for (int j = 0; j < NT_PER_SPLIT; ++j) {
        const int v0 = (tile0 + j) * 32;
        const unsigned bufSel = (unsigned)(j & 1);

        // issue async loads for tile j+1 into the other buffer
        if (j + 1 < NT_PER_SPLIT) {
            const unsigned short* src = ebf + (size_t)(v0 + 32) * D_DIM;
            const unsigned obuf = (bufSel ^ 1u) * (unsigned)BUF_BYTE;
            #pragma unroll
            for (int c = 0; c < CHUNKS; ++c)
                async_b128(lbyte[c] + obuf, src + goff[c]);
            wait_async_le12();  // tile j's 12 loads done (in-order completion)
        } else {
            wait_async_0();
        }
        __syncthreads();        // buffer (j&1) fully populated for all waves

        // ---- two 16-column subtiles; 24 WMMAs each, dual accumulator chains.
        //      contiguous-K B layout (same k order as A): lane n: col=n, K=k0+0..15; lane n+16: K+16..31
        #pragma unroll
        for (int s = 0; s < 2; ++s) {
            const unsigned short* bp =
                &Blds[bufSel][((s << 4) + nlane) * LDBP + half * 16];
            v8f acc0 = {}, acc1 = {};
            FragU bcur, bnxt;
            bcur.u[0] = *(const u32x4*)(bp);
            bcur.u[1] = *(const u32x4*)(bp + 8);
            #pragma unroll
            for (int kb = 0; kb < KB; ++kb) {
                if (kb + 1 < KB) {
                    bnxt.u[0] = *(const u32x4*)(bp + (kb + 1) * 32);
                    bnxt.u[1] = *(const u32x4*)(bp + (kb + 1) * 32 + 8);
                }
                if ((kb & 1) == 0)
                    acc0 = __builtin_amdgcn_wmma_f32_16x16x32_bf16(
                        false, afrag[kb], false, bcur.v, (short)0, acc0,
                        false, false);
                else
                    acc1 = __builtin_amdgcn_wmma_f32_16x16x32_bf16(
                        false, afrag[kb], false, bcur.v, (short)0, acc1,
                        false, false);
                bcur = bnxt;
            }

            // score = x.e - 0.5*||e||^2 ; update running argmax.
            // C layout: elem i, lane l -> row = i + 8*(l>=16), col = l&15
            const int  cand = v0 + (s << 4) + nlane;
            const float he  = halfe2[cand];
            #pragma unroll
            for (int i = 0; i < 8; ++i) {
                float sc = (acc0[i] + acc1[i]) - he;
                if (sc > bestv[i]) { bestv[i] = sc; besti[i] = cand; }
            }
        }

        __syncthreads();        // all reads of buf (j&1) done before j+2 lands
    }

    // ---- reduce across the 16 lanes holding the same 8 rows (xor within half)
    #pragma unroll
    for (int m = 1; m < 16; m <<= 1) {
        #pragma unroll
        for (int i = 0; i < 8; ++i) {
            float ov = __shfl_xor(bestv[i], m, 32);
            int   oi = __shfl_xor(besti[i], m, 32);
            if (ov > bestv[i] || (ov == bestv[i] && oi < besti[i])) {
                bestv[i] = ov; besti[i] = oi;
            }
        }
    }

    if (nlane == 0) {
        #pragma unroll
        for (int i = 0; i < 8; ++i) {
            int row = aRow0 + half * 8 + i;
            partVal[row * SPLIT + split] = bestv[i];
            partIdx[row * SPLIT + split] = besti[i];
        }
    }
}

// ---------------------------------------------------------------------------
// Kernel 4: combine V-split partials -> nearest vocabulary index per row
// ---------------------------------------------------------------------------
__global__ void combine_kernel(const float* __restrict__ partVal,
                               const int* __restrict__ partIdx,
                               int* __restrict__ nearest) {
    int r = blockIdx.x * blockDim.x + threadIdx.x;
    if (r >= NROWS) return;
    float bv = -3.4e38f; int bi = 0;
    #pragma unroll
    for (int s = 0; s < SPLIT; ++s) {
        float v = partVal[r * SPLIT + s];
        int   i = partIdx[r * SPLIT + s];
        if (v > bv || (v == bv && i < bi)) { bv = v; bi = i; }
    }
    nearest[r] = bi;
}

// ---------------------------------------------------------------------------
// Kernel 5: out[cls, 0, :]      = prefix
//           out[cls, 1..16, :]  = token_embedding[nearest]  (f32, exact gather)
//           out[cls, 17..76, :] = suffix
// ---------------------------------------------------------------------------
__global__ void assemble_kernel(const float* __restrict__ prefix,
                                const float* __restrict__ suffix,
                                const float* __restrict__ emb,
                                const int* __restrict__ nearest,
                                float* __restrict__ out, long long total) {
    long long idx = (long long)blockIdx.x * blockDim.x + threadIdx.x;
    if (idx >= total) return;
    int d    = (int)(idx % D_DIM);
    int rest = (int)(idx / D_DIM);
    int pos  = rest % SEQ_LEN;
    int cls  = rest / SEQ_LEN;
    float v;
    if (pos == 0) {
        v = prefix[(size_t)cls * D_DIM + d];
    } else if (pos <= N_CTX) {
        int nn = nearest[cls * N_CTX + (pos - 1)];
        v = emb[(size_t)nn * D_DIM + d];
    } else {
        v = suffix[((size_t)cls * SUF_LEN + (pos - 1 - N_CTX)) * D_DIM + d];
    }
    out[idx] = v;
}

// ---------------------------------------------------------------------------
extern "C" void kernel_launch(void* const* d_in, const int* in_sizes, int n_in,
                              void* d_out, int out_size, void* d_ws, size_t ws_size,
                              hipStream_t stream) {
    const float* ctx    = (const float*)d_in[0];
    const float* emb    = (const float*)d_in[1];
    const float* prefix = (const float*)d_in[2];
    const float* suffix = (const float*)d_in[3];
    float* out = (float*)d_out;

    // workspace layout (~90 MB)
    char*  ws  = (char*)d_ws;
    size_t off = 0;
    auto alloc = [&](size_t bytes) -> void* {
        void* p = ws + off;
        off += (bytes + 255) & ~(size_t)255;
        return p;
    };
    unsigned short* xbf     = (unsigned short*)alloc((size_t)NROWS * D_DIM * 2);
    unsigned short* ebf     = (unsigned short*)alloc((size_t)V_DIM * D_DIM * 2);
    float*          halfe2  = (float*)alloc((size_t)V_DIM * 4);
    float*          partVal = (float*)alloc((size_t)NROWS * SPLIT * 4);
    int*            partIdx = (int*)alloc((size_t)NROWS * SPLIT * 4);
    int*            nearest = (int*)alloc((size_t)NROWS * 4);
    (void)ws_size; (void)in_sizes; (void)n_in; (void)out_size;

    // 1) ctx -> bf16
    {
        int n = NROWS * D_DIM;
        cvt_x_kernel<<<(n + 255) / 256, 256, 0, stream>>>(ctx, xbf, n);
    }
    // 2) embedding -> bf16 + 0.5*||e||^2
    cvt_e_kernel<<<V_DIM, 256, 0, stream>>>(emb, ebf, halfe2);

    // 3) fused GEMM + argmax (the WMMA workhorse)
    {
        dim3 grid(NROWS / M_PER_WG, SPLIT);   // (64, 8)
        gemm_argmax_kernel<<<grid, 256, 0, stream>>>(xbf, ebf, halfe2,
                                                     partVal, partIdx);
    }
    // 4) combine splits
    combine_kernel<<<(NROWS + 255) / 256, 256, 0, stream>>>(partVal, partIdx,
                                                            nearest);
    // 5) assemble [512, 77, 768] output
    {
        long long total = (long long)N_CLS * SEQ_LEN * D_DIM;
        int blocks = (int)((total + 255) / 256);
        assemble_kernel<<<blocks, 256, 0, stream>>>(prefix, suffix, emb,
                                                    nearest, out, total);
    }
}
